// MambaEncoder_2525440770256
// MI455X (gfx1250) — matrix operand order repaired and checked by reading, not compiled, not run on hardware
//
#include <hip/hip_runtime.h>
#include <hip/hip_bf16.h>
#include <math.h>

// ---------------------------------------------------------------------------
// Model dims (fixed by reference setup_inputs)
// ---------------------------------------------------------------------------
#define BSZ    4
#define LSEQ   2048
#define DMODEL 1024
#define DI     2048          // d_inner
#define DSTATE 16
#define DTR    64
#define NLAYER 2
#define MROWS  (BSZ * LSEQ)  // 8192

// GEMM tiling: block = 256 threads = 8 waves; wave computes 32 rows x 64 cols
#define BM 256
#define BN 64
#define BK 32
#define SA_ELEMS (BM * BK)   // 8192 bf16
#define SB_ELEMS (BN * BK)   // 2048 bf16

typedef __attribute__((ext_vector_type(16))) __bf16 v16bf;
typedef __attribute__((ext_vector_type(8)))  float  v8f;

union Frag { uint4 q[2]; v16bf v; };

__device__ __forceinline__ float silu_f(float x)     { return x / (1.f + __expf(-x)); }
__device__ __forceinline__ float gelu_f(float x)     { return 0.5f * x * (1.f + erff(x * 0.70710678118654752f)); }
__device__ __forceinline__ float softplus_f(float x) { return (x > 20.f) ? x : log1pf(__expf(x)); }

// CDNA5 async global->LDS copy (ASYNCcnt-tracked). vdst = LDS byte offset.
__device__ __forceinline__ void async_copy_b128(unsigned lds_off, const void* gaddr) {
    asm volatile("global_load_async_to_lds_b128 %0, %1, off"
                 :: "v"(lds_off), "v"(gaddr) : "memory");
}

// ---------------------------------------------------------------------------
// Fused GEMM:  C[M,N] = act( A[M,K] * W[N,K]^T + bias )
//   A, W are bf16 (pre-converted). Outputs: Cf (f32, optional) and/or
//   Cbf (bf16, optional, feeds the next GEMM).
//   act: 0 = none, 1 = bias+softplus, 2 = bias+gelu(exact)
// Double-buffered LDS fed by global_load_async_to_lds_b128; per K-tile each
// wave issues 12 ds_load_b128 up front then 8 back-to-back WMMAs
// (2 A-frags x 4 B-frags, B reused across A-frags, disjoint accumulators).
// ---------------------------------------------------------------------------
__global__ __launch_bounds__(256) void gemm_bf16_wmma(
    const __bf16* __restrict__ A, int lda,
    const __bf16* __restrict__ W,
    const float*  __restrict__ bias,
    float*        __restrict__ Cf,
    __bf16*       __restrict__ Cbf,
    int M, int N, int K, int act)
{
    // [2 bufs A | 2 bufs B] = 40 KB LDS
    __shared__ __align__(16) __bf16 lds[2 * SA_ELEMS + 2 * SB_ELEMS];

    const int tid  = threadIdx.x;
    const int wave = tid >> 5;
    const int lane = tid & 31;
    const int l16  = lane & 15;
    const int hi   = lane >> 4;           // 0: lanes 0-15, 1: lanes 16-31
    const int bm   = blockIdx.x * BM;
    const int bn   = blockIdx.y * BN;
    const int mrow = wave << 5;           // 32 rows per wave

    auto sa = [&](int buf, int row, int k) -> int { return buf * SA_ELEMS + row * BK + k; };
    auto sb = [&](int buf, int row, int k) -> int { return 2 * SA_ELEMS + buf * SB_ELEMS + row * BK + k; };

    // Issue one K-tile's async copies: 4 A-chunks + 1 B-chunk per thread
    // => ASYNCcnt += 5 per wave per stage (uniform, no divergence).
    auto stage = [&](int kt, int buf) {
        #pragma unroll
        for (int it = 0; it < 4; ++it) {
            int idx = tid + it * 256;          // 0..1023
            int row = idx >> 2;                // 0..255
            int ck  = (idx & 3) << 3;          // 0,8,16,24 (bf16 elems)
            const __bf16* g = A + (size_t)(bm + row) * lda + kt + ck;
            unsigned l = (unsigned)(uintptr_t)&lds[sa(buf, row, ck)];
            async_copy_b128(l, g);
        }
        {
            int row  = tid >> 2;               // 0..63
            int ck   = (tid & 3) << 3;
            int wrow = bn + row;
            if (wrow >= N) wrow = N - 1;       // clamp; cols >= N never stored
            const __bf16* g = W + (size_t)wrow * K + kt + ck;
            unsigned l = (unsigned)(uintptr_t)&lds[sb(buf, row, ck)];
            async_copy_b128(l, g);
        }
    };

    v8f acc0[4] = {{}, {}, {}, {}};
    v8f acc1[4] = {{}, {}, {}, {}};

    // Load all fragments first (12x ds_load_b128 -> single dscnt wait),
    // then 8 WMMAs back-to-back (no D->A/B hazard: disjoint accumulators).
    auto compute = [&](int buf) {
        Frag a0, a1, b[4];
        a0.q[0] = *(const uint4*)&lds[sa(buf, mrow + l16,      hi * 8)];
        a0.q[1] = *(const uint4*)&lds[sa(buf, mrow + l16,      hi * 8 + 16)];
        a1.q[0] = *(const uint4*)&lds[sa(buf, mrow + 16 + l16, hi * 8)];
        a1.q[1] = *(const uint4*)&lds[sa(buf, mrow + 16 + l16, hi * 8 + 16)];
        #pragma unroll
        for (int ns = 0; ns < 4; ++ns) {
            b[ns].q[0] = *(const uint4*)&lds[sb(buf, ns * 16 + l16, hi * 8)];
            b[ns].q[1] = *(const uint4*)&lds[sb(buf, ns * 16 + l16, hi * 8 + 16)];
        }
        #pragma unroll
        for (int ns = 0; ns < 4; ++ns) {
            acc0[ns] = __builtin_amdgcn_wmma_f32_16x16x32_bf16(
                false, a0.v, false, b[ns].v, (short)0, acc0[ns], false, false);
            acc1[ns] = __builtin_amdgcn_wmma_f32_16x16x32_bf16(
                false, a1.v, false, b[ns].v, (short)0, acc1[ns], false, false);
        }
    };

    const int T = K / BK;
    stage(0, 0);

    // steady state: overlap next tile's async copies with this tile's WMMAs
    int t = 0;
    for (; t < T - 1; ++t) {
        stage((t + 1) * BK, (t + 1) & 1);
        asm volatile("s_wait_asynccnt 0x5" ::: "memory");  // tile t complete
        __syncthreads();
        compute(t & 1);
        __syncthreads();   // all waves done reading buf before it is restaged
    }
    // tail tile
    asm volatile("s_wait_asynccnt 0x0" ::: "memory");
    __syncthreads();
    compute(t & 1);

    // Epilogue. C/D layout: lane -> column, VGPR j -> row (j / j+8 halves).
    #pragma unroll
    for (int h = 0; h < 2; ++h) {
        #pragma unroll
        for (int ns = 0; ns < 4; ++ns) {
            const int col = bn + ns * 16 + l16;
            if (col >= N) continue;
            const float bv = (act != 0 && bias != nullptr) ? bias[col] : 0.f;
            #pragma unroll
            for (int j = 0; j < 8; ++j) {
                const int grow = bm + mrow + h * 16 + (hi << 3) + j;
                float v = h ? acc1[ns][j] : acc0[ns][j];
                if (act == 1)      v = softplus_f(v + bv);
                else if (act == 2) v = gelu_f(v + bv);
                if (Cf)  Cf [(size_t)grow * N + col] = v;
                if (Cbf) Cbf[(size_t)grow * N + col] = (__bf16)v;
            }
        }
    }
}

// ---------------------------------------------------------------------------
// f32 -> bf16 bulk conversion (n multiple of 8); native v_cvt path
// ---------------------------------------------------------------------------
__global__ __launch_bounds__(256) void cvt_f32_to_bf16(
    const float* __restrict__ s, __bf16* __restrict__ d, long n)
{
    long i = ((long)blockIdx.x * 256 + threadIdx.x) * 8;
    if (i >= n) return;
    const float4 a = *(const float4*)(s + i);
    const float4 b = *(const float4*)(s + i + 4);
    union { __bf16 h[8]; uint4 q; } u;
    u.h[0] = (__bf16)a.x; u.h[1] = (__bf16)a.y; u.h[2] = (__bf16)a.z; u.h[3] = (__bf16)a.w;
    u.h[4] = (__bf16)b.x; u.h[5] = (__bf16)b.y; u.h[6] = (__bf16)b.z; u.h[7] = (__bf16)b.w;
    *(uint4*)(d + i) = u.q;
}

// ---------------------------------------------------------------------------
// Depthwise causal conv1d (width 4) + bias + SiLU; emits f32 (scan u) + bf16
// ---------------------------------------------------------------------------
__global__ __launch_bounds__(256) void conv_silu_kernel(
    const float* __restrict__ xz,   // [B,L,2*DI], x_in = [..., :DI]
    const float* __restrict__ cw,   // [DI,4]
    const float* __restrict__ cb,   // [DI]
    float*  __restrict__ xc,
    __bf16* __restrict__ xcbf)
{
    const size_t idx = (size_t)blockIdx.x * blockDim.x + threadIdx.x;
    if (idx >= (size_t)BSZ * LSEQ * DI) return;
    const int d = (int)(idx % DI);
    const int l = (int)((idx / DI) % LSEQ);
    const int b = (int)(idx / ((size_t)DI * LSEQ));

    float acc = cb[d];
    #pragma unroll
    for (int j = 0; j < 4; ++j) {
        const int lj = l - 3 + j;
        if (lj >= 0)
            acc += cw[d * 4 + j] * xz[((size_t)b * LSEQ + lj) * (2 * DI) + d];
    }
    const float v = silu_f(acc);
    xc[idx]   = v;
    xcbf[idx] = (__bf16)v;
}

// ---------------------------------------------------------------------------
// Selective scan fused with y *= silu(z); emits bf16 y for out_proj GEMM.
// 16-lane group per (b,d) channel; lane = state index; shuffle-reduce.
// ---------------------------------------------------------------------------
__global__ __launch_bounds__(256) void scan_kernel(
    const float* __restrict__ xc,    // u  [B,L,DI]
    const float* __restrict__ dtb,   // dt [B,L,DI] (post-softplus)
    const float* __restrict__ xdbl,  // [B,L,96]; B at +64, C at +80
    const float* __restrict__ xz,    // z at xz[...,DI:2*DI]
    const float* __restrict__ Alog,  // [DI,16]
    const float* __restrict__ Dp,    // [DI]
    __bf16* __restrict__ ybf)        // [B,L,DI]
{
    const int s   = threadIdx.x & 15;
    const int grp = blockIdx.x * (blockDim.x >> 4) + (threadIdx.x >> 4);
    if (grp >= BSZ * DI) return;
    const int b = grp / DI;
    const int d = grp % DI;

    const float Av = -__expf(Alog[(size_t)d * DSTATE + s]);
    const float Dv = Dp[d];

    const float* Bp  = xdbl + (size_t)b * LSEQ * 96 + 64 + s;
    const float* Cp  = Bp + 16;
    const float* up  = xc  + (size_t)b * LSEQ * DI + d;
    const float* dtp = dtb + (size_t)b * LSEQ * DI + d;
    const float* zp  = xz  + (size_t)b * LSEQ * (2 * DI) + DI + d;
    __bf16*      yp  = ybf + (size_t)b * LSEQ * DI + d;

    float h = 0.f;
    for (int t = 0; t < LSEQ; ++t) {
        const float dtv = dtp[(size_t)t * DI];
        const float uv  = up [(size_t)t * DI];
        const float Bv  = Bp [(size_t)t * 96];
        const float Cv  = Cp [(size_t)t * 96];
        h = __expf(dtv * Av) * h + dtv * Bv * uv;
        float part = h * Cv;
        #pragma unroll
        for (int off = 8; off >= 1; off >>= 1)
            part += __shfl_xor(part, off, 16);
        if (s == 0) {
            const float zv = zp[(size_t)t * (2 * DI)];
            yp[(size_t)t * DI] = (__bf16)((part + uv * Dv) * silu_f(zv));
        }
    }
}

// ---------------------------------------------------------------------------
// out = rmsnorm(xa + xb) * w ; optional bf16 mirror for the next GEMM input
// ---------------------------------------------------------------------------
__global__ __launch_bounds__(256) void add_rmsnorm_kernel(
    const float* __restrict__ xa,
    const float* __restrict__ xb,
    const float* __restrict__ w,
    float*  __restrict__ out,
    __bf16* __restrict__ outbf)
{
    __shared__ float red[8];
    __shared__ float rinv;
    const size_t row = blockIdx.x;
    const float* pa = xa + row * DMODEL;
    const float* pb = xb + row * DMODEL;

    float vals[4];
    float s = 0.f;
    #pragma unroll
    for (int i = 0; i < 4; ++i) {
        const int c = threadIdx.x + i * 256;
        const float v = pa[c] + pb[c];
        vals[i] = v;
        s += v * v;
    }
    #pragma unroll
    for (int off = 16; off >= 1; off >>= 1) s += __shfl_xor(s, off, 32);
    if ((threadIdx.x & 31) == 0) red[threadIdx.x >> 5] = s;
    __syncthreads();
    if (threadIdx.x == 0) {
        float tot = 0.f;
        #pragma unroll
        for (int i = 0; i < 8; ++i) tot += red[i];
        rinv = rsqrtf(tot / (float)DMODEL + 1.1920929e-07f);
    }
    __syncthreads();
    const float r = rinv;
    #pragma unroll
    for (int i = 0; i < 4; ++i) {
        const int c = threadIdx.x + i * 256;
        const float v = vals[i] * r * w[c];
        out[row * DMODEL + c] = v;
        if (outbf) outbf[row * DMODEL + c] = (__bf16)v;
    }
}

// ---------------------------------------------------------------------------
// Orchestration
// ---------------------------------------------------------------------------
extern "C" void kernel_launch(void* const* d_in, const int* in_sizes, int n_in,
                              void* d_out, int out_size, void* d_ws, size_t ws_size,
                              hipStream_t stream) {
    const float* x_in      = (const float*)d_in[0];
    const float* in_proj_w = (const float*)d_in[1];   // [2, 4096, 1024]
    const float* conv_w    = (const float*)d_in[2];   // [2, 2048, 4]
    const float* conv_b    = (const float*)d_in[3];   // [2, 2048]
    const float* x_proj_w  = (const float*)d_in[4];   // [2, 96, 2048]
    const float* dt_proj_w = (const float*)d_in[5];   // [2, 2048, 64]
    const float* dt_proj_b = (const float*)d_in[6];   // [2, 2048]
    const float* A_log     = (const float*)d_in[7];   // [2, 2048, 16]
    const float* D_param   = (const float*)d_in[8];   // [2, 2048]
    const float* out_proj_w= (const float*)d_in[9];   // [2, 1024, 2048]
    const float* mlp_w1    = (const float*)d_in[10];  // [2048, 1024]
    const float* mlp_b1    = (const float*)d_in[11];
    const float* mlp_w2    = (const float*)d_in[12];  // [1024, 2048]
    const float* mlp_b2    = (const float*)d_in[13];
    const float* ln_w      = (const float*)d_in[14];
    const float* ln2_w     = (const float*)d_in[15];

    // ---- workspace layout: f32 region then bf16 region (~510 MB total)
    float* xz   = (float*)d_ws;                       // [8192, 4096]
    float* xc   = xz   + (size_t)MROWS * 4096;        // [8192, 2048]
    float* xdbl = xc   + (size_t)MROWS * 2048;        // [8192, 96]
    float* dtb  = xdbl + (size_t)MROWS * 96;          // [8192, 2048]
    float* yb   = dtb  + (size_t)MROWS * 2048;        // [8192, 1024] (mlp2 out)
    float* xcur = yb   + (size_t)MROWS * 1024;        // [8192, 1024]

    __bf16* xbf    = (__bf16*)(xcur + (size_t)MROWS * 1024);
    __bf16* xcbf   = xbf    + (size_t)MROWS * 1024;   // [8192, 2048]
    __bf16* xdblbf = xcbf   + (size_t)MROWS * 2048;   // [8192, 96]
    __bf16* ybf    = xdblbf + (size_t)MROWS * 96;     // [8192, 2048]
    __bf16* t1bf   = ybf    + (size_t)MROWS * 2048;   // [8192, 1024]
    __bf16* h1bf   = t1bf   + (size_t)MROWS * 1024;   // [8192, 2048]
    __bf16* ipw_bf = h1bf   + (size_t)MROWS * 2048;   // [2,4096,1024]
    __bf16* xpw_bf = ipw_bf + (size_t)2 * 4096 * 1024;
    __bf16* dpw_bf = xpw_bf + (size_t)2 * 96 * DI;
    __bf16* opw_bf = dpw_bf + (size_t)2 * DI * DTR;
    __bf16* w1bf   = opw_bf + (size_t)2 * DMODEL * DI;
    __bf16* w2bf   = w1bf   + (size_t)2 * DMODEL * DMODEL;

    auto cvt = [&](const float* s, __bf16* d, long n) {
        cvt_f32_to_bf16<<<(unsigned)((n / 8 + 255) / 256), 256, 0, stream>>>(s, d, n);
    };

    // one-time conversions: activations entry point + all weights
    cvt(x_in,      xbf,    (long)MROWS * DMODEL);
    cvt(in_proj_w, ipw_bf, (long)2 * 4096 * 1024);
    cvt(x_proj_w,  xpw_bf, (long)2 * 96 * DI);
    cvt(dt_proj_w, dpw_bf, (long)2 * DI * DTR);
    cvt(out_proj_w,opw_bf, (long)2 * DMODEL * DI);
    cvt(mlp_w1,    w1bf,   (long)2 * DMODEL * DMODEL);
    cvt(mlp_w2,    w2bf,   (long)2 * DMODEL * DMODEL);

    hipMemcpyAsync(xcur, x_in, (size_t)MROWS * DMODEL * sizeof(float),
                   hipMemcpyDeviceToDevice, stream);

    auto gemm = [&](const __bf16* A, int lda, const __bf16* W, const float* bias,
                    float* Cf, __bf16* Cbf, int M, int N, int K, int act) {
        dim3 g(M / BM, (N + BN - 1) / BN);
        gemm_bf16_wmma<<<g, 256, 0, stream>>>(A, lda, W, bias, Cf, Cbf, M, N, K, act);
    };

    for (int i = 0; i < NLAYER; ++i) {
        const __bf16* ipw = ipw_bf + (size_t)i * 4096 * 1024;
        const float*  cwp = conv_w + (size_t)i * DI * 4;
        const float*  cbp = conv_b + (size_t)i * DI;
        const __bf16* xpw = xpw_bf + (size_t)i * 96 * DI;
        const __bf16* dpw = dpw_bf + (size_t)i * DI * DTR;
        const float*  dpb = dt_proj_b + (size_t)i * DI;
        const float*  alp = A_log  + (size_t)i * DI * DSTATE;
        const float*  dpp = D_param+ (size_t)i * DI;
        const __bf16* opw = opw_bf + (size_t)i * DMODEL * DI;

        // 1) in_proj: xz = x @ in_w^T                     [8192, 4096] f32
        gemm(xbf, DMODEL, ipw, nullptr, xz, nullptr, MROWS, 2 * DI, DMODEL, 0);

        // 2) depthwise causal conv + SiLU -> xc (f32+bf16)
        conv_silu_kernel<<<((size_t)MROWS * DI) / 256, 256, 0, stream>>>(
            xz, cwp, cbp, xc, xcbf);

        // 3) x_proj: x_dbl = xc @ xp_w^T                  [8192, 96] f32+bf16
        gemm(xcbf, DI, xpw, nullptr, xdbl, xdblbf, MROWS, DTR + 2 * DSTATE, DI, 0);

        // 4) dt = softplus(x_dbl[:, :64] @ dtp_w^T + b)   [8192, 2048] f32
        gemm(xdblbf, DTR + 2 * DSTATE, dpw, dpb, dtb, nullptr, MROWS, DI, DTR, 1);

        // 5) selective scan + fused y *= silu(z) -> bf16  [8192, 2048]
        scan_kernel<<<(BSZ * DI) / 16, 256, 0, stream>>>(
            xc, dtb, xdbl, xz, alp, dpp, ybf);

        // 6) out_proj -> t1 (bf16 only)                   [8192, 1024]
        gemm(ybf, DI, opw, nullptr, nullptr, t1bf, MROWS, DMODEL, DI, 0);

        // 7) MLP1: h1 = gelu(t1 @ w1^T + b1) (bf16 only)  [8192, 2048]
        gemm(t1bf, DMODEL, w1bf, mlp_b1, nullptr, h1bf, MROWS, 2 * DMODEL, DMODEL, 2);

        // 8) MLP2: yb = gelu(h1 @ w2^T + b2) (f32 only)   [8192, 1024]
        gemm(h1bf, 2 * DMODEL, w2bf, mlp_b2, yb, nullptr, MROWS, DMODEL, 2 * DMODEL, 2);

        // 9) x = rmsnorm(mlp_out + residual) * ln_w  (xcur f32 + xbf bf16)
        add_rmsnorm_kernel<<<MROWS, 256, 0, stream>>>(yb, xcur, ln_w, xcur, xbf);
    }

    // final: out = rmsnorm(x + original_input) * ln2_w
    add_rmsnorm_kernel<<<MROWS, 256, 0, stream>>>(xcur, x_in, ln2_w, (float*)d_out, nullptr);
}
